// ScaledDotProductAttentionMemory_4672924418743
// MI455X (gfx1250) — compile-verified, matrix operand order checked by reading, not run on hardware
//
#include <hip/hip_runtime.h>
#include <hip/hip_bf16.h>
#include <math.h>

typedef __attribute__((ext_vector_type(16))) _Float16 v16h;
typedef __attribute__((ext_vector_type(8)))  _Float16 v8h;
typedef __attribute__((ext_vector_type(2)))  __fp16   v2hk;   // cvt_pkrtz result type
typedef __attribute__((ext_vector_type(4)))  __fp16   v4hk;
typedef __attribute__((ext_vector_type(8)))  float    v8f;
typedef __attribute__((ext_vector_type(4)))  float    v4f;

typedef __attribute__((ext_vector_type(4))) unsigned int tdm_g0_t;
typedef __attribute__((ext_vector_type(8))) int          tdm_g1_t;
typedef __attribute__((ext_vector_type(4))) int          tdm_g2_t;

#define D_MODEL 1024
#define HD      1024      // H * D_K = H * D_V
#define HEADS   16
#define DK      64
#define BATCH   2
#define NQ      2048
#define NK      2048
#define MROWS   (BATCH * NQ)   // 4096 rows for all GEMMs

// Q pre-scale: (1/sqrt(64)) * log2(e); softmax then runs in the exp2 domain.
#define QSCALE  0.1803368801111204f

// ---------------------------------------------------------------------------
__device__ __forceinline__ void store_pk_f16(void* dst, float a, float b) {
    *(v2hk*)dst = __builtin_amdgcn_cvt_pkrtz(a, b);
}

// ---------------------------------------------------------------------------
// VALU 16-lane xor-shuffle via v_permlane16_b32 (no LDS traffic).
// ---------------------------------------------------------------------------
__device__ __forceinline__ float xor_lane16(float v, unsigned s0, unsigned s1) {
    unsigned u = __builtin_bit_cast(unsigned, v);
    u = (unsigned)__builtin_amdgcn_permlane16(u, u, s0, s1, false, false);
    return __builtin_bit_cast(float, u);
}
__device__ __forceinline__ float red_max16(float v) {
    v = fmaxf(v, xor_lane16(v, 0x67452301u, 0xEFCDAB89u));   // xor 1
    v = fmaxf(v, xor_lane16(v, 0x54761032u, 0xDCFE98BAu));   // xor 2
    v = fmaxf(v, xor_lane16(v, 0x32107654u, 0xBA98FEDCu));   // xor 4
    v = fmaxf(v, xor_lane16(v, 0xFEDCBA98u, 0x76543210u));   // xor 8
    return v;
}
__device__ __forceinline__ float red_sum16(float v) {
    v += xor_lane16(v, 0x67452301u, 0xEFCDAB89u);
    v += xor_lane16(v, 0x54761032u, 0xDCFE98BAu);
    v += xor_lane16(v, 0x32107654u, 0xBA98FEDCu);
    v += xor_lane16(v, 0xFEDCBA98u, 0x76543210u);
    return v;
}

// ---------------------------------------------------------------------------
__device__ __forceinline__ unsigned lds_off(const void* p) {
    return (unsigned)(unsigned long long)(__attribute__((address_space(3))) const char*)p;
}

// ---------------------------------------------------------------------------
// Tensor Data Mover: async 2D f16 tile load, global -> LDS, with LDS row
// padding.  padIntervalCode: pad after 2^(c+1) DWORDs; padAmountCode: (c+1) DW.
// ---------------------------------------------------------------------------
__device__ __forceinline__ void tdm_load_tile_f16(
    const _Float16* gsrc, unsigned ldsByteOff,
    unsigned tileX, unsigned tileY, unsigned rowStrideElems,
    unsigned padIntervalCode, unsigned padAmountCode)
{
    unsigned long long ga = (unsigned long long)(const void*)gsrc;
    tdm_g0_t g0;
    g0[0] = 1u;                                           // count=1 (valid), user mode
    g0[1] = ldsByteOff;                                   // lds_addr
    g0[2] = (unsigned)ga;                                 // global_addr[31:0]
    g0[3] = ((unsigned)(ga >> 32) & 0x01FFFFFFu) | (2u << 30);  // addr[56:32] | type=2
    tdm_g1_t g1;
    g1[0] = (int)((1u << 16)                              // data_size = 2 bytes
                | (1u << 20)                              // pad_enable
                | (padIntervalCode << 22)
                | (padAmountCode << 25));
    g1[1] = (int)((tileX & 0xFFFFu) << 16);               // tensor_dim0[15:0]
    g1[2] = (int)(((tileX >> 16) & 0xFFFFu) | ((tileY & 0xFFFFu) << 16));
    g1[3] = (int)(((tileY >> 16) & 0xFFFFu) | ((tileX & 0xFFFFu) << 16)); // | tile_dim0
    g1[4] = (int)(tileY & 0xFFFFu);                       // tile_dim1 (tile_dim2 = 0)
    g1[5] = (int)rowStrideElems;                          // tensor_dim0_stride[31:0]
    g1[6] = 0;
    g1[7] = 0;
    tdm_g2_t g2 = {0, 0, 0, 0};
    tdm_g2_t g3 = {0, 0, 0, 0};
#if defined(__clang_major__) && (__clang_major__ >= 23)
    tdm_g1_t g4 = {0, 0, 0, 0, 0, 0, 0, 0};
    __builtin_amdgcn_tensor_load_to_lds(g0, g1, g2, g3, g4, 0);
#else
    __builtin_amdgcn_tensor_load_to_lds(g0, g1, g2, g3, 0);
#endif
}

// ---------------------------------------------------------------------------
// 16x32 f16 A/B fragment from a row-major tile (ISA 7.12.2 layout).
// ---------------------------------------------------------------------------
__device__ __forceinline__ v16h frag_ld(const _Float16* rowptr, int kbase) {
    v8h lo = *(const v8h*)(rowptr + kbase);
    v8h hi = *(const v8h*)(rowptr + kbase + 16);
    v16h f;
#pragma unroll
    for (int i = 0; i < 8; ++i) { f[i] = lo[i]; f[i + 8] = hi[i]; }
    return f;
}

// Fragment gather when the K dimension is strided (row-major V tile).
__device__ __forceinline__ v16h frag_ld_strided(const _Float16* base, int ld,
                                                int col, int kbase) {
    v16h f;
#pragma unroll
    for (int i = 0; i < 8; ++i) {
        f[i]     = base[(kbase + i) * ld + col];
        f[i + 8] = base[(kbase + 16 + i) * ld + col];
    }
    return f;
}

__device__ __forceinline__ v8f wmma16(v16h a, v16h b, v8f c) {
    return __builtin_amdgcn_wmma_f32_16x16x32_f16(
        false, a, false, b, (short)0, c, false, false);
}

// ===========================================================================
// Weight prep: W_f32 (K x N row-major) -> Wt_f16 (N x K row-major).
// ===========================================================================
__global__ __launch_bounds__(256) void transpose_w_kernel(
    const float* __restrict__ W, _Float16* __restrict__ Wt)
{
    __shared__ float t[32][33];
    const int tid = threadIdx.x;
    const int n0 = blockIdx.x * 32;
    const int k0 = blockIdx.y * 32;
#pragma unroll
    for (int i = 0; i < 4; ++i) {
        int idx = tid + i * 256;
        int r = idx >> 5, c = idx & 31;
        t[r][c] = W[(size_t)(k0 + r) * D_MODEL + n0 + c];
    }
    __syncthreads();
#pragma unroll
    for (int i = 0; i < 2; ++i) {
        int p = tid + i * 256;
        int r = p >> 4;
        int c = (p & 15) << 1;
        store_pk_f16(&Wt[(size_t)(n0 + r) * D_MODEL + k0 + c], t[c][r], t[c + 1][r]);
    }
}

// ===========================================================================
// Projection GEMM with double-buffered TDM B tiles (prefetch 1 iter ahead).
// SCALEQ folds the attention scale * log2(e) into the Q projection.
// ===========================================================================
template <int GATE, int SCALEQ>
__global__ __launch_bounds__(256) void proj_gemm_kernel(
    const float* __restrict__ A, const _Float16* __restrict__ Wt,
    const float* __restrict__ bias,
    const float* __restrict__ mem, const float* __restrict__ Wmm,
    const float* __restrict__ bmm,
    _Float16* __restrict__ out)
{
    __shared__ _Float16 As[128][40];      // staged with packed cvt
    __shared__ _Float16 Bs[2][128][40];   // TDM double buffer

    const int tid   = threadIdx.x;
    const int lane  = tid & 31;
    const int wave  = tid >> 5;
    const int bn    = blockIdx.x;
    const int bm    = blockIdx.y;
    const int waveM = wave >> 2;
    const int waveN = wave & 3;
    const int kb16  = (lane >> 4) << 3;
    const int l15   = lane & 15;
    const unsigned bsOff[2] = { lds_off(&Bs[0][0][0]), lds_off(&Bs[1][0][0]) };

    v8f zero = {};
    v8f acc[4][2];
#pragma unroll
    for (int i = 0; i < 4; ++i)
#pragma unroll
        for (int j = 0; j < 2; ++j) acc[i][j] = zero;

    const int NIT = D_MODEL / 32;
    if (wave == 0) {
        tdm_load_tile_f16(Wt + (size_t)(bn * 128) * D_MODEL, bsOff[0],
                          32, 128, D_MODEL, 3, 3);
    }
    for (int it = 0; it < NIT; ++it) {
        const int k0 = it * 32;
        const bool hasNext = (it + 1) < NIT;
        if (wave == 0 && hasNext) {
            tdm_load_tile_f16(Wt + (size_t)(bn * 128) * D_MODEL + k0 + 32,
                              bsOff[(it + 1) & 1], 32, 128, D_MODEL, 3, 3);
        }
        // ---- A tile 128x32: float4 loads, 2x pkrtz, 64-bit LDS stores ----
#pragma unroll
        for (int i = 0; i < 4; ++i) {
            int q = tid + i * 256;
            int r = q >> 3;
            int c = (q & 7) << 2;
            v4f x = *(const v4f*)&A[(size_t)(bm * 128 + r) * D_MODEL + k0 + c];
            v2hk lo = __builtin_amdgcn_cvt_pkrtz(x.x, x.y);
            v2hk hi = __builtin_amdgcn_cvt_pkrtz(x.z, x.w);
            v4hk o4; o4[0] = lo[0]; o4[1] = lo[1]; o4[2] = hi[0]; o4[3] = hi[1];
            *(v4hk*)&As[r][c] = o4;
        }
        if (wave == 0) {
            if (hasNext) __builtin_amdgcn_s_wait_tensorcnt(1);
            else         __builtin_amdgcn_s_wait_tensorcnt(0);
        }
        __syncthreads();

        const int cur = it & 1;
        v16h af[4], bf[2];
#pragma unroll
        for (int mi = 0; mi < 4; ++mi)
            af[mi] = frag_ld(&As[waveM * 64 + mi * 16 + l15][0], kb16);
#pragma unroll
        for (int ni = 0; ni < 2; ++ni)
            bf[ni] = frag_ld(&Bs[cur][waveN * 32 + ni * 16 + l15][0], kb16);
#pragma unroll
        for (int mi = 0; mi < 4; ++mi)
#pragma unroll
            for (int ni = 0; ni < 2; ++ni)
                acc[mi][ni] = wmma16(af[mi], bf[ni], acc[mi][ni]);
        __syncthreads();
    }

#pragma unroll
    for (int mi = 0; mi < 4; ++mi) {
#pragma unroll
        for (int ni = 0; ni < 2; ++ni) {
#pragma unroll
            for (int r = 0; r < 8; ++r) {
                int row = bm * 128 + waveM * 64 + mi * 16 + r + ((lane >> 4) << 3);
                int col = bn * 128 + waveN * 32 + ni * 16 + l15;
                float v = acc[mi][ni][r] + bias[col];
                if (GATE)   v *= (mem[row] * Wmm[col] + bmm[col]);
                if (SCALEQ) v *= QSCALE;
                out[(size_t)row * HD + col] = (_Float16)v;
            }
        }
    }
}

// ===========================================================================
// Flash attention: double-buffered TDM K/V tile pairs, exp2-domain softmax,
// permlane16 reductions.  Q is pre-scaled by 0.125*log2(e) at projection.
// ===========================================================================
__global__ __launch_bounds__(256) void attn_kernel(
    const _Float16* __restrict__ qh, const _Float16* __restrict__ kh,
    const _Float16* __restrict__ vh, _Float16* __restrict__ oh)
{
    __shared__ _Float16 Ks[2][32][72];    // [key][d] + 8-half pad (TDM)
    __shared__ _Float16 Vs[2][32][72];    // [key][d] row-major    (TDM)
    __shared__ _Float16 Ps[8][16][40];    // per-wave P relayout buffer

    const int tid  = threadIdx.x;
    const int lane = tid & 31;
    const int wave = tid >> 5;
    const int qtiles = NQ / 128;
    int bid = blockIdx.x;
    const int qt = bid % qtiles;  bid /= qtiles;
    const int h  = bid % HEADS;   bid /= HEADS;
    const int b  = bid;
    const int q0   = qt * 128 + wave * 16;
    const int kb16 = (lane >> 4) << 3;
    const int l15  = lane & 15;
    const unsigned ksOff[2] = { lds_off(&Ks[0][0][0]), lds_off(&Ks[1][0][0]) };
    const unsigned vsOff[2] = { lds_off(&Vs[0][0][0]), lds_off(&Vs[1][0][0]) };

    v16h qa[2];
#pragma unroll
    for (int c = 0; c < 2; ++c) {
        const _Float16* p = qh + (size_t)(b * NQ + q0 + l15) * HD + h * 64 + c * 32;
        qa[c] = frag_ld(p, kb16);
    }

    v8f zero = {};
    v8f o[4];
#pragma unroll
    for (int d = 0; d < 4; ++d) o[d] = zero;
    float mrow[8], lrow[8];
#pragma unroll
    for (int r = 0; r < 8; ++r) { mrow[r] = -1e30f; lrow[r] = 0.f; }

    const int NIT = NK / 32;
    const _Float16* kBase = kh + (size_t)(b * NK) * HD + h * 64;
    const _Float16* vBase = vh + (size_t)(b * NK) * HD + h * 64;
    if (wave == 0) {
        tdm_load_tile_f16(kBase, ksOff[0], 64, 32, HD, 4, 3);
        tdm_load_tile_f16(vBase, vsOff[0], 64, 32, HD, 4, 3);
    }
    for (int it = 0; it < NIT; ++it) {
        const bool hasNext = (it + 1) < NIT;
        if (wave == 0) {
            if (hasNext) {
                const size_t nofs = (size_t)(it + 1) * 32 * HD;
                tdm_load_tile_f16(kBase + nofs, ksOff[(it + 1) & 1], 64, 32, HD, 4, 3);
                tdm_load_tile_f16(vBase + nofs, vsOff[(it + 1) & 1], 64, 32, HD, 4, 3);
                __builtin_amdgcn_s_wait_tensorcnt(2);
            } else {
                __builtin_amdgcn_s_wait_tensorcnt(0);
            }
        }
        __syncthreads();
        const int cur = it & 1;

        // ---- S = Q Ktr (already exp2-scaled via Q) ----
        v8f s0 = zero, s1 = zero;
#pragma unroll
        for (int c = 0; c < 2; ++c) {
            v16h b0 = frag_ld(&Ks[cur][l15][c * 32], kb16);
            v16h b1 = frag_ld(&Ks[cur][16 + l15][c * 32], kb16);
            s0 = wmma16(qa[c], b0, s0);
            s1 = wmma16(qa[c], b1, s1);
        }

        // ---- online softmax (base-2 domain) ----
#pragma unroll
        for (int r = 0; r < 8; ++r) {
            float a0 = s0[r], a1 = s1[r];
            float mx = red_max16(fmaxf(a0, a1));
            float mnew  = fmaxf(mrow[r], mx);
            float alpha = __builtin_amdgcn_exp2f(mrow[r] - mnew);
            float p0 = __builtin_amdgcn_exp2f(a0 - mnew);
            float p1 = __builtin_amdgcn_exp2f(a1 - mnew);
            float rs = red_sum16(p0 + p1);
            lrow[r] = lrow[r] * alpha + rs;
            mrow[r] = mnew;
#pragma unroll
            for (int d = 0; d < 4; ++d) o[d][r] *= alpha;

            int prow = r + ((lane >> 4) << 3);
            Ps[wave][prow][l15]      = (_Float16)p0;
            Ps[wave][prow][16 + l15] = (_Float16)p1;
        }
        // wave-private LDS round trip (C-layout -> A-layout)
        asm volatile("s_wait_dscnt 0" ::: "memory");

        // ---- O += P V ----
        v16h pa = frag_ld(&Ps[wave][l15][0], kb16);
#pragma unroll
        for (int d = 0; d < 4; ++d) {
            v16h vf = frag_ld_strided(&Vs[cur][0][0], 72, d * 16 + l15, kb16);
            o[d] = wmma16(pa, vf, o[d]);
        }
        __syncthreads();
    }

#pragma unroll
    for (int d = 0; d < 4; ++d) {
#pragma unroll
        for (int r = 0; r < 8; ++r) {
            int row = q0 + r + ((lane >> 4) << 3);
            int col = h * 64 + d * 16 + l15;
            oh[(size_t)(b * NQ + row) * HD + col] = (_Float16)(o[d][r] / lrow[r]);
        }
    }
}

// ===========================================================================
// Output GEMM: both tiles via double-buffered TDM; loop body is pure
// TDM-prefetch + fragment loads + WMMA.
// ===========================================================================
__global__ __launch_bounds__(256) void out_gemm_kernel(
    const _Float16* __restrict__ A, const _Float16* __restrict__ Wt,
    const float* __restrict__ bias, float* __restrict__ out)
{
    __shared__ _Float16 As[2][128][40];
    __shared__ _Float16 Bs[2][128][40];

    const int tid   = threadIdx.x;
    const int lane  = tid & 31;
    const int wave  = tid >> 5;
    const int bn    = blockIdx.x;
    const int bm    = blockIdx.y;
    const int waveM = wave >> 2;
    const int waveN = wave & 3;
    const int kb16  = (lane >> 4) << 3;
    const int l15   = lane & 15;
    const unsigned asOff[2] = { lds_off(&As[0][0][0]), lds_off(&As[1][0][0]) };
    const unsigned bsOff[2] = { lds_off(&Bs[0][0][0]), lds_off(&Bs[1][0][0]) };

    v8f zero = {};
    v8f acc[4][2];
#pragma unroll
    for (int i = 0; i < 4; ++i)
#pragma unroll
        for (int j = 0; j < 2; ++j) acc[i][j] = zero;

    const int NIT = HD / 32;
    const _Float16* aBase = A  + (size_t)(bm * 128) * HD;
    const _Float16* bBase = Wt + (size_t)(bn * 128) * D_MODEL;
    if (wave == 0) {
        tdm_load_tile_f16(aBase, asOff[0], 32, 128, HD, 3, 3);
        tdm_load_tile_f16(bBase, bsOff[0], 32, 128, D_MODEL, 3, 3);
    }
    for (int it = 0; it < NIT; ++it) {
        const bool hasNext = (it + 1) < NIT;
        if (wave == 0) {
            if (hasNext) {
                tdm_load_tile_f16(aBase + (it + 1) * 32, asOff[(it + 1) & 1],
                                  32, 128, HD, 3, 3);
                tdm_load_tile_f16(bBase + (it + 1) * 32, bsOff[(it + 1) & 1],
                                  32, 128, D_MODEL, 3, 3);
                __builtin_amdgcn_s_wait_tensorcnt(2);
            } else {
                __builtin_amdgcn_s_wait_tensorcnt(0);
            }
        }
        __syncthreads();
        const int cur = it & 1;

        v16h af[4], bf[2];
#pragma unroll
        for (int mi = 0; mi < 4; ++mi)
            af[mi] = frag_ld(&As[cur][waveM * 64 + mi * 16 + l15][0], kb16);
#pragma unroll
        for (int ni = 0; ni < 2; ++ni)
            bf[ni] = frag_ld(&Bs[cur][waveN * 32 + ni * 16 + l15][0], kb16);
#pragma unroll
        for (int mi = 0; mi < 4; ++mi)
#pragma unroll
            for (int ni = 0; ni < 2; ++ni)
                acc[mi][ni] = wmma16(af[mi], bf[ni], acc[mi][ni]);
        __syncthreads();
    }

#pragma unroll
    for (int mi = 0; mi < 4; ++mi) {
#pragma unroll
        for (int ni = 0; ni < 2; ++ni) {
#pragma unroll
            for (int r = 0; r < 8; ++r) {
                int row = bm * 128 + waveM * 64 + mi * 16 + r + ((lane >> 4) << 3);
                int col = bn * 128 + waveN * 32 + ni * 16 + l15;
                out[(size_t)row * D_MODEL + col] = acc[mi][ni][r] + bias[col];
            }
        }
    }
}

// ===========================================================================
extern "C" void kernel_launch(void* const* d_in, const int* in_sizes, int n_in,
                              void* d_out, int out_size, void* d_ws, size_t ws_size,
                              hipStream_t stream) {
    (void)in_sizes; (void)n_in; (void)out_size; (void)ws_size;
    const float* queries = (const float*)d_in[0];
    const float* keys    = (const float*)d_in[1];
    const float* values  = (const float*)d_in[2];
    const float* memory  = (const float*)d_in[3];
    const float* Wq  = (const float*)d_in[4];
    const float* bq  = (const float*)d_in[5];
    const float* Wk  = (const float*)d_in[6];
    const float* bk  = (const float*)d_in[7];
    const float* Wv  = (const float*)d_in[8];
    const float* bv  = (const float*)d_in[9];
    const float* Wo  = (const float*)d_in[10];
    const float* bo  = (const float*)d_in[11];
    const float* Wmm = (const float*)d_in[12];
    const float* bmm = (const float*)d_in[13];
    float* out = (float*)d_out;

    const size_t actBytes = (size_t)MROWS * HD * sizeof(_Float16);        // 8 MB
    const size_t wBytes   = (size_t)D_MODEL * D_MODEL * sizeof(_Float16); // 2 MB
    char* ws = (char*)d_ws;
    _Float16* qh  = (_Float16*)(ws);
    _Float16* kh  = (_Float16*)(ws + actBytes);
    _Float16* vh  = (_Float16*)(ws + 2 * actBytes);
    _Float16* oh  = (_Float16*)(ws + 3 * actBytes);
    _Float16* WtQ = (_Float16*)(ws + 4 * actBytes);
    _Float16* WtK = (_Float16*)(ws + 4 * actBytes + wBytes);
    _Float16* WtV = (_Float16*)(ws + 4 * actBytes + 2 * wBytes);
    _Float16* WtO = (_Float16*)(ws + 4 * actBytes + 3 * wBytes);

    dim3 gt(D_MODEL / 32, D_MODEL / 32);
    transpose_w_kernel<<<gt, 256, 0, stream>>>(Wq, WtQ);
    transpose_w_kernel<<<gt, 256, 0, stream>>>(Wk, WtK);
    transpose_w_kernel<<<gt, 256, 0, stream>>>(Wv, WtV);
    transpose_w_kernel<<<gt, 256, 0, stream>>>(Wo, WtO);

    dim3 gg(HD / 128, MROWS / 128);
    proj_gemm_kernel<0, 1><<<gg, 256, 0, stream>>>(queries, WtQ, bq, memory, Wmm, bmm, qh);
    proj_gemm_kernel<1, 0><<<gg, 256, 0, stream>>>(keys,    WtK, bk, memory, Wmm, bmm, kh);
    proj_gemm_kernel<1, 0><<<gg, 256, 0, stream>>>(values,  WtV, bv, memory, Wmm, bmm, vh);

    attn_kernel<<<BATCH * HEADS * (NQ / 128), 256, 0, stream>>>(qh, kh, vh, oh);

    out_gemm_kernel<<<gg, 256, 0, stream>>>(oh, WtO, bo, out);
}
